// GCNModel_1005022347672
// MI455X (gfx1250) — compile-verified, matrix operand order checked by reading, not run on hardware
//
#include <hip/hip_runtime.h>
#include <hip/hip_bf16.h>

// ---------------------------------------------------------------------------
// GCN 4-layer pipeline for gfx1250 (MI455X), wave32, WMMA bf16 GEMMs.
// GEMM v2: barrier-free, LDS-free; W pre-packed into WMMA B-fragment layout.
// ---------------------------------------------------------------------------

#define N_NODES 50000
#define N_EDGES 800000
#define D_IN    128
#define D_H     256
#define D_OUT   128
#define BN_EPS  1e-5f
#define LRELU_SLOPE 0.01f

typedef __bf16 v8bf  __attribute__((ext_vector_type(8)));
typedef __bf16 v16bf __attribute__((ext_vector_type(16)));
typedef float  v8f   __attribute__((ext_vector_type(8)));

__device__ __forceinline__ void atomicAddF(float* p, float v) {
    __hip_atomic_fetch_add(p, v, __ATOMIC_RELAXED, __HIP_MEMORY_SCOPE_AGENT);
}

// ---------------------------------------------------------------------------
// Degree computation: deg[i] = 1 + #incoming edges; then nisq = rsqrt(deg)
// ---------------------------------------------------------------------------
__global__ void deg_init_kernel(float* deg, int n) {
    int i = blockIdx.x * 256 + threadIdx.x;
    if (i < n) deg[i] = 1.0f;
}

__global__ void deg_scatter_kernel(const int* __restrict__ dst, float* deg, int e) {
    int i = blockIdx.x * 256 + threadIdx.x;
    if (i < e) atomicAddF(&deg[dst[i]], 1.0f);
}

__global__ void deg_rsqrt_kernel(float* deg, int n) {
    int i = blockIdx.x * 256 + threadIdx.x;
    if (i < n) deg[i] = rsqrtf(deg[i]);
}

// ---------------------------------------------------------------------------
// fp32 -> bf16 conversion (activations feeding WMMA)
// ---------------------------------------------------------------------------
__global__ void f2bf_kernel(const float* __restrict__ in, __bf16* __restrict__ out,
                            unsigned n) {
    unsigned i = blockIdx.x * 256u + threadIdx.x;
    if (i < n) out[i] = (__bf16)in[i];
}

// ---------------------------------------------------------------------------
// Pack W[di x dout] (fp32, row-major) into WMMA B-fragment order (bf16):
//   Wpk[((ct*nkt + kt)*32 + lane)*16 + e] = bf16( W[K][col] )
//     K   = kt*32 + ((e>>3)<<4) + ((lane>>4)<<3) + (e&7)   (ISA 16-bit striping)
//     col = ct*16 + (lane&15)
// One thread per packed element; done once per layer, reused by 3125 row tiles.
// ---------------------------------------------------------------------------
__global__ void pack_w_kernel(const float* __restrict__ W, __bf16* __restrict__ Wpk,
                              int di, int dout) {
    unsigned idx = blockIdx.x * 256u + threadIdx.x;
    unsigned total = (unsigned)di * (unsigned)dout;
    if (idx >= total) return;
    int e    = idx & 15;
    int lane = (idx >> 4) & 31;
    int nkt  = di >> 5;
    int kt   = (int)((idx >> 9) % (unsigned)nkt);
    int ct   = (int)(idx / ((unsigned)nkt << 9));
    int K    = kt * 32 + ((e >> 3) << 4) + ((lane >> 4) << 3) + (e & 7);
    int col  = ct * 16 + (lane & 15);
    Wpk[idx] = (__bf16)W[(size_t)K * dout + col];
}

// ---------------------------------------------------------------------------
// GEMM: C[N x dout] = A[N x di] @ W[di x dout], bf16 in, f32 accumulate.
// Block = 256 threads = 8 waves; no LDS, no barriers.
// Wave w owns output rows [blockIdx.x*16, +16) and TPW col tiles
// (cols w*TPW*16 .. ), TPW = dout/128. Per K-step: A fragment = two
// contiguous 16B global loads per lane (ISA A 16x32 striping), B fragment =
// one contiguous 32B load from the packed weights; A reused across TPW WMMAs.
// EXEC stays all-ones: N = 3125*16 exactly, dims % 128 == 0.
// ---------------------------------------------------------------------------
template <int TPW>
__global__ __launch_bounds__(256) void gemm_bf16_kernel(
    const __bf16* __restrict__ A,    // [N, di] row-major
    const __bf16* __restrict__ Wpk,  // packed B fragments
    float* __restrict__ C,           // [N, dout]
    int di, int dout)
{
    const int wave = threadIdx.x >> 5;
    const int lane = threadIdx.x & 31;
    const int M    = lane & 15;      // A row / B,C col within tile
    const int half = lane >> 4;      // K sub-band select
    const int rowbase = blockIdx.x * 16;
    const int nkt = di >> 5;

    v8f acc[TPW] = {};

    const __bf16* arow = A + (size_t)(rowbase + M) * di + half * 8;

    for (int kt = 0; kt < nkt; ++kt) {
        v8bf a0 = *(const v8bf*)(arow + kt * 32);
        v8bf a1 = *(const v8bf*)(arow + kt * 32 + 16);
        v16bf av = __builtin_shufflevector(a0, a1,
                     0,1,2,3,4,5,6,7,8,9,10,11,12,13,14,15);
        if (kt + 1 < nkt)   // gfx1250 global_prefetch_b8 of next K tile
            __builtin_prefetch(arow + (kt + 1) * 32, 0, 0);
        #pragma unroll
        for (int j = 0; j < TPW; ++j) {
            int ct = wave * TPW + j;
            v16bf bv = *(const v16bf*)(Wpk +
                         (((size_t)ct * nkt + kt) * 32 + lane) * 16);
            acc[j] = __builtin_amdgcn_wmma_f32_16x16x32_bf16(
                         false, av, false, bv, (short)0, acc[j], false, false);
        }
    }

    // C/D layout: VGPR r -> M = r + 8*half, N = lane&15
    #pragma unroll
    for (int j = 0; j < TPW; ++j) {
        int col = (wave * TPW + j) * 16 + M;
        #pragma unroll
        for (int r = 0; r < 8; ++r)
            C[(size_t)(rowbase + r + 8 * half) * dout + col] = acc[j][r];
    }
}

// ---------------------------------------------------------------------------
// Self-loop + bias: agg = xl * nisq^2 + b  (initializes agg buffer)
// ---------------------------------------------------------------------------
__global__ void selfloop_kernel(const float* __restrict__ xl,
                                const float* __restrict__ nisq,
                                const float* __restrict__ bias,
                                float* __restrict__ agg, int n, int dout) {
    unsigned i = blockIdx.x * 256u + threadIdx.x;
    unsigned total = (unsigned)n * (unsigned)dout;
    if (i >= total) return;
    int row = i / dout;
    int c   = i - row * dout;
    float ns = nisq[row];
    agg[i] = xl[i] * ns * ns + bias[c];
}

// ---------------------------------------------------------------------------
// Edge scatter: one wave per edge; lanes stride channels.
// Coalesced 128B gather of xl[src], global_atomic_add_f32 into agg[dst].
// ---------------------------------------------------------------------------
__global__ __launch_bounds__(256) void edge_scatter_kernel(
    const int* __restrict__ src, const int* __restrict__ dst,
    const float* __restrict__ nisq, const float* __restrict__ xl,
    float* __restrict__ agg, int e, int dout)
{
    int edge = blockIdx.x * 8 + (threadIdx.x >> 5);
    int lane = threadIdx.x & 31;
    if (edge >= e) return;
    int s = src[edge], d = dst[edge];
    float coef = nisq[s] * nisq[d];
    const float* xrow = xl + (size_t)s * dout;
    float* arow = agg + (size_t)d * dout;
    for (int c = lane; c < dout; c += 32)
        atomicAddF(&arow[c], xrow[c] * coef);
}

// ---------------------------------------------------------------------------
// Residual add for layer 4: agg += x
// ---------------------------------------------------------------------------
__global__ void resid_kernel(float* __restrict__ agg, const float* __restrict__ x,
                             unsigned n) {
    unsigned i = blockIdx.x * 256u + threadIdx.x;
    if (i < n) agg[i] += x[i];
}

// ---------------------------------------------------------------------------
// BatchNorm: column stats (sum + sumsq), finalize, apply.
// ---------------------------------------------------------------------------
__global__ void zero_stats_kernel(float* sum, float* sumsq, int dout) {
    int t = threadIdx.x;
    if (t < dout) { sum[t] = 0.f; sumsq[t] = 0.f; }
}

__global__ __launch_bounds__(256) void bn_stats_kernel(
    const float* __restrict__ x, float* sum, float* sumsq, int n, int dout)
{
    int c = threadIdx.x;
    if (c >= dout) return;
    float s = 0.f, sq = 0.f;
    for (int r = blockIdx.x; r < n; r += gridDim.x) {
        float v = x[(size_t)r * dout + c];
        s += v; sq += v * v;
    }
    atomicAddF(&sum[c], s);
    atomicAddF(&sumsq[c], sq);
}

__global__ void bn_final_kernel(const float* sum, const float* sumsq,
                                float* mean, float* inv, int n, int dout) {
    int c = threadIdx.x;
    if (c >= dout) return;
    float m = sum[c] / (float)n;
    float v = sumsq[c] / (float)n - m * m;   // biased variance
    mean[c] = m;
    inv[c]  = rsqrtf(v + BN_EPS);
}

// BN apply (+optional LeakyReLU), in-place; optionally emits bf16 for next GEMM.
__global__ void bn_apply_kernel(float* __restrict__ x,
                                const float* __restrict__ mean,
                                const float* __restrict__ inv,
                                const float* __restrict__ g,
                                const float* __restrict__ be,
                                __bf16* __restrict__ obf,
                                int n, int dout, int do_lrelu) {
    unsigned i = blockIdx.x * 256u + threadIdx.x;
    unsigned total = (unsigned)n * (unsigned)dout;
    if (i >= total) return;
    int c = i % dout;
    float v = (x[i] - mean[c]) * inv[c] * g[c] + be[c];
    if (do_lrelu && v < 0.f) v *= LRELU_SLOPE;
    x[i] = v;
    if (obf) obf[i] = (__bf16)v;
}

// ---------------------------------------------------------------------------
// LayerNorm over D_OUT=128 channels: one wave per row, wave32 shuffle reduce.
// ---------------------------------------------------------------------------
__global__ __launch_bounds__(256) void ln_kernel(
    const float* __restrict__ x, const float* __restrict__ g,
    const float* __restrict__ b, float* __restrict__ out, int n)
{
    int row  = blockIdx.x * 8 + (threadIdx.x >> 5);
    int lane = threadIdx.x & 31;
    if (row >= n) return;
    const float* xr = x + (size_t)row * D_OUT;
    float v[4];
    float s = 0.f;
    #pragma unroll
    for (int j = 0; j < 4; ++j) { v[j] = xr[lane + 32 * j]; s += v[j]; }
    #pragma unroll
    for (int off = 16; off; off >>= 1) s += __shfl_xor(s, off, 32);
    float m = s * (1.f / D_OUT);
    float vs = 0.f;
    #pragma unroll
    for (int j = 0; j < 4; ++j) { float d = v[j] - m; vs += d * d; }
    #pragma unroll
    for (int off = 16; off; off >>= 1) vs += __shfl_xor(vs, off, 32);
    float rinv = rsqrtf(vs * (1.f / D_OUT) + BN_EPS);
    float* orow = out + (size_t)row * D_OUT;
    #pragma unroll
    for (int j = 0; j < 4; ++j) {
        int c = lane + 32 * j;
        orow[c] = (v[j] - m) * rinv * g[c] + b[c];
    }
}

// ---------------------------------------------------------------------------
// Host orchestration
// ---------------------------------------------------------------------------
extern "C" void kernel_launch(void* const* d_in, const int* in_sizes, int n_in,
                              void* d_out, int out_size, void* d_ws, size_t ws_size,
                              hipStream_t stream) {
    (void)in_sizes; (void)n_in; (void)out_size; (void)ws_size;

    const float* x   = (const float*)d_in[0];
    const int*   ei  = (const int*)d_in[1];
    const int*   src = ei;
    const int*   dst = ei + N_EDGES;

    const float* Wc[4]  = { (const float*)d_in[2],  (const float*)d_in[6],
                            (const float*)d_in[10], (const float*)d_in[14] };
    const float* bc[4]  = { (const float*)d_in[3],  (const float*)d_in[7],
                            (const float*)d_in[11], (const float*)d_in[15] };
    const float* gc[4]  = { (const float*)d_in[4],  (const float*)d_in[8],
                            (const float*)d_in[12], (const float*)d_in[16] };
    const float* bec[4] = { (const float*)d_in[5],  (const float*)d_in[9],
                            (const float*)d_in[13], (const float*)d_in[17] };
    const float* ln_g = (const float*)d_in[18];
    const float* ln_b = (const float*)d_in[19];
    float* out = (float*)d_out;

    const int din[4] = { D_IN, D_H, D_H, D_H };
    const int dou[4] = { D_H,  D_H, D_H, D_OUT };

    // Workspace carve-up (256B aligned): ~128.4 MB total.
    char* base = (char*)d_ws;
    size_t off = 0;
    auto carve = [&](size_t bytes) -> char* {
        char* p = base + off;
        off = (off + bytes + 255) & ~(size_t)255;
        return p;
    };
    float*  nisq  = (float*)carve((size_t)N_NODES * 4);
    float*  xl    = (float*)carve((size_t)N_NODES * D_H * 4);
    float*  agg   = (float*)carve((size_t)N_NODES * D_H * 4);
    __bf16* actbf = (__bf16*)carve((size_t)N_NODES * D_H * 2);
    __bf16* wpk   = (__bf16*)carve((size_t)D_H * D_H * 2);
    float*  ssum  = (float*)carve(256 * 4);
    float*  ssq   = (float*)carve(256 * 4);
    float*  smean = (float*)carve(256 * 4);
    float*  sinv  = (float*)carve(256 * 4);

    // 1) degrees -> nisq
    deg_init_kernel<<<(N_NODES + 255) / 256, 256, 0, stream>>>(nisq, N_NODES);
    deg_scatter_kernel<<<(N_EDGES + 255) / 256, 256, 0, stream>>>(dst, nisq, N_EDGES);
    deg_rsqrt_kernel<<<(N_NODES + 255) / 256, 256, 0, stream>>>(nisq, N_NODES);

    // 2) input activations to bf16
    f2bf_kernel<<<((unsigned)N_NODES * D_IN + 255) / 256, 256, 0, stream>>>(
        x, actbf, (unsigned)N_NODES * D_IN);

    // 3) four GCN layers
    for (int l = 0; l < 4; ++l) {
        const int di = din[l], dO = dou[l];
        const unsigned nd = (unsigned)N_NODES * (unsigned)dO;

        // pack + convert weights into WMMA B-fragment order
        pack_w_kernel<<<((unsigned)(di * dO) + 255) / 256, 256, 0, stream>>>(
            Wc[l], wpk, di, dO);

        if (dO == 256)
            gemm_bf16_kernel<2><<<N_NODES / 16, 256, 0, stream>>>(
                actbf, wpk, xl, di, dO);
        else
            gemm_bf16_kernel<1><<<N_NODES / 16, 256, 0, stream>>>(
                actbf, wpk, xl, di, dO);

        selfloop_kernel<<<(nd + 255) / 256, 256, 0, stream>>>(
            xl, nisq, bc[l], agg, N_NODES, dO);
        edge_scatter_kernel<<<(N_EDGES + 7) / 8, 256, 0, stream>>>(
            src, dst, nisq, xl, agg, N_EDGES, dO);

        if (l == 3)  // residual: h4 + identity before BN4
            resid_kernel<<<(nd + 255) / 256, 256, 0, stream>>>(agg, x, nd);

        zero_stats_kernel<<<1, 256, 0, stream>>>(ssum, ssq, dO);
        bn_stats_kernel<<<256, 256, 0, stream>>>(agg, ssum, ssq, N_NODES, dO);
        bn_final_kernel<<<1, 256, 0, stream>>>(ssum, ssq, smean, sinv, N_NODES, dO);
        bn_apply_kernel<<<(nd + 255) / 256, 256, 0, stream>>>(
            agg, smean, sinv, gc[l], bec[l],
            (l < 3) ? actbf : (__bf16*)nullptr, N_NODES, dO, (l < 3) ? 1 : 0);
    }

    // 4) final LayerNorm -> d_out
    ln_kernel<<<(N_NODES + 7) / 8, 256, 0, stream>>>(agg, ln_g, ln_b, out, N_NODES);
}
